// GatedCrossCAFM_40218073760038
// MI455X (gfx1250) — matrix operand order repaired and checked by reading, not compiled, not run on hardware
//
#include <hip/hip_runtime.h>
#include <hip/hip_bf16.h>
#include <stdint.h>

#define BB 4
#define CC 256
#define HH 64
#define WW 64
#define NN (HH*WW)   // 4096
#define CR 64        // C/R

#ifndef __has_builtin
#define __has_builtin(x) 0
#endif

#if __has_builtin(__builtin_amdgcn_global_load_async_to_lds_b128)
#define HAS_ASYNC 1
#else
#define HAS_ASYNC 0
#endif

#if HAS_ASYNC
# if __has_builtin(__builtin_amdgcn_s_wait_asynccnt)
#  define WAIT_ASYNC(n) __builtin_amdgcn_s_wait_asynccnt(n)
# else
#  define WAIT_ASYNC(n) asm volatile("s_wait_asynccnt %0" :: "i"(n) : "memory")
# endif
#else
# define WAIT_ASYNC(n) ((void)0)
#endif

#if __has_builtin(__builtin_amdgcn_s_wait_dscnt)
# define WAIT_DS0() __builtin_amdgcn_s_wait_dscnt(0)
#else
# define WAIT_DS0() asm volatile("s_wait_dscnt 0x0" ::: "memory")
#endif

typedef __bf16 bf16_t;
typedef __attribute__((ext_vector_type(16))) __bf16 bf16x16;
typedef __attribute__((ext_vector_type(8)))  float  floatx8;
typedef __attribute__((ext_vector_type(4)))  unsigned int uintx4;
typedef __attribute__((ext_vector_type(4)))  int intx4;
typedef __attribute__((address_space(1))) intx4* gv4_p;   // global v4i32*
typedef __attribute__((address_space(3))) intx4* lv4_p;   // LDS v4i32*

static __device__ __forceinline__ unsigned short f32_to_bf16(float f) {
  unsigned int u = __float_as_uint(f);
  u += 0x7fffu + ((u >> 16) & 1u);
  return (unsigned short)(u >> 16);
}

// A-fragment (16x32, M x K): lane sub = row M; half 0 -> K {0..7,16..23}, half 1 -> K {8..15,24..31}
static __device__ __forceinline__ bf16x16 load_frag_a(const unsigned short* rowptr, int half) {
  union { uintx4 u[2]; bf16x16 v; } f;
  f.u[0] = *reinterpret_cast<const uintx4*>(rowptr + half * 8);
  f.u[1] = *reinterpret_cast<const uintx4*>(rowptr + 16 + half * 8);
  return f.v;
}

// B-fragment (32x16, K x N): lane sub = column N; half 0 -> K {0..7,16..23}, half 1 -> K {8..15,24..31}
static __device__ __forceinline__ bf16x16 load_frag_b(const unsigned short* colptr, int half) {
  union { uintx4 u[2]; bf16x16 v; } f;
  f.u[0] = *reinterpret_cast<const uintx4*>(colptr + half * 8);
  f.u[1] = *reinterpret_cast<const uintx4*>(colptr + 16 + half * 8);
  return f.v;
}

static __device__ __forceinline__ floatx8 wmma_bf16(bf16x16 a, bf16x16 b, floatx8 c) {
  return __builtin_amdgcn_wmma_f32_16x16x32_bf16(false, a, false, b, (short)0, c, false, false);
}

// 16-byte global -> LDS copy: async (ASYNCcnt) when available, else through VGPRs.
static __device__ __forceinline__ void cp16_to_lds(unsigned short* lds, const unsigned short* g) {
#if HAS_ASYNC
  __builtin_amdgcn_global_load_async_to_lds_b128(
      (gv4_p)(void*)g, (lv4_p)(void*)lds, 0, 0);
#else
  *reinterpret_cast<uintx4*>(lds) = *reinterpret_cast<const uintx4*>(g);
#endif
}

// ---------------------------------------------------------------------------
// (B,C,N) f32  ->  (B,N,C) bf16   LDS tiled transpose
__global__ void k_transpose_bf16(const float* __restrict__ src, unsigned short* __restrict__ dst) {
  __shared__ float tile[32][33];
  int b  = blockIdx.z;
  int c0 = blockIdx.x * 32;
  int n0 = blockIdx.y * 32;
  int tx = threadIdx.x;   // 32
  int ty = threadIdx.y;   // 8
  #pragma unroll
  for (int r = 0; r < 4; ++r) {
    int c = c0 + ty + r * 8;
    tile[ty + r * 8][tx] = src[((size_t)b * CC + c) * NN + n0 + tx];
  }
  __syncthreads();
  #pragma unroll
  for (int r = 0; r < 4; ++r) {
    int n = n0 + ty + r * 8;
    dst[((size_t)b * NN + n) * CC + c0 + tx] = f32_to_bf16(tile[tx][ty + r * 8]);
  }
}

__global__ void k_convert_bf16(const float* __restrict__ src, unsigned short* __restrict__ dst, int n) {
  int i = blockIdx.x * 256 + threadIdx.x;
  if (i < n) dst[i] = f32_to_bf16(src[i]);
}

// ---------------------------------------------------------------------------
// gate[b] = sigmoid(g2( relu(g1( mean_hw(x_self) )) ))
__global__ void k_gate(const float* __restrict__ xs,
                       const float* __restrict__ g1w, const float* __restrict__ g1b,
                       const float* __restrict__ g2w, const float* __restrict__ g2b,
                       float* __restrict__ gate) {
  __shared__ float p[CC];
  __shared__ float h[CR];
  int b = blockIdx.x, t = threadIdx.x;
  const float4* row = reinterpret_cast<const float4*>(xs + ((size_t)b * CC + t) * NN);
  float s = 0.f;
  for (int i = 0; i < NN / 4; ++i) { float4 v = row[i]; s += v.x + v.y + v.z + v.w; }
  p[t] = s * (1.0f / NN);
  __syncthreads();
  if (t < CR) {
    float a = g1b[t];
    for (int c = 0; c < CC; ++c) a += g1w[t * CC + c] * p[c];
    h[t] = fmaxf(a, 0.f);
  }
  __syncthreads();
  if (t == 0) {
    float z = g2b[0];
    for (int j = 0; j < CR; ++j) z += g2w[j] * h[j];
    gate[b] = 1.f / (1.f + __expf(-z));
  }
}

// ---------------------------------------------------------------------------
// depthwise 3x3 (SAME) + bias + BN + SiLU  -> local (B,C,H,W) f32
__global__ void k_dwconv(const float* __restrict__ xs,
                         const float* __restrict__ dww, const float* __restrict__ dwb,
                         const float* __restrict__ bng, const float* __restrict__ bnb,
                         const float* __restrict__ bnm, const float* __restrict__ bnv,
                         float* __restrict__ local) {
  size_t idx = (size_t)blockIdx.x * 256 + threadIdx.x;     // over B*C*H*W
  int w = (int)(idx & (WW - 1));
  int h = (int)((idx >> 6) & (HH - 1));
  int c = (int)((idx >> 12) & (CC - 1));
  const float* plane = xs + ((idx >> 12) << 12);
  float acc = 0.f;
  #pragma unroll
  for (int dy = -1; dy <= 1; ++dy)
    #pragma unroll
    for (int dx = -1; dx <= 1; ++dx) {
      int hh = h + dy, ww = w + dx;
      if (hh >= 0 && hh < HH && ww >= 0 && ww < WW)
        acc += dww[c * 9 + (dy + 1) * 3 + (dx + 1)] * plane[hh * WW + ww];
    }
  acc += dwb[c];
  float sc = bng[c] * rsqrtf(bnv[c] + 1e-5f);
  float y = (acc - bnm[c]) * sc + bnb[c];
  local[idx] = y * (1.f / (1.f + __expf(-y)));
}

// ---------------------------------------------------------------------------
// Q projection: q_t(B,N,C) bf16 = [Wq (C,C) @ X (C,N)]^T   via WMMA
__global__ void k_proj_q(const unsigned short* __restrict__ xt,
                         const unsigned short* __restrict__ wq,
                         const float* __restrict__ qb,
                         unsigned short* __restrict__ qt) {
  int wave = (int)((blockIdx.x * 256 + threadIdx.x) >> 5);
  int lane = threadIdx.x & 31;
  int half = lane >> 4, sub = lane & 15;
  int nt = wave & 255, ot = (wave >> 8) & 15, b = wave >> 12;
  int o0 = ot * 16, n0 = nt * 16;
  floatx8 acc;
  #pragma unroll
  for (int i = 0; i < 8; ++i) acc[i] = 0.f;
  const unsigned short* arow = wq + (size_t)(o0 + sub) * CC;
  const unsigned short* bcol = xt + ((size_t)b * NN + n0 + sub) * CC;
  #pragma unroll
  for (int kk = 0; kk < CC; kk += 32)
    acc = wmma_bf16(load_frag_a(arow + kk, half), load_frag_b(bcol + kk, half), acc);
  int n = n0 + sub;
  union { unsigned short s[8]; uintx4 u; } ou;
  #pragma unroll
  for (int r = 0; r < 8; ++r) ou.s[r] = f32_to_bf16(acc[r] + qb[o0 + 8 * half + r]);
  *reinterpret_cast<uintx4*>(qt + ((size_t)b * NN + n) * CC + o0 + 8 * half) = ou.u;
}

// KV projection: first 256 rows -> k_t (B,N,C) bf16 (transposed), last 256 -> v (B,C,N) bf16
__global__ void k_proj_kv(const unsigned short* __restrict__ xt,
                          const unsigned short* __restrict__ wkv,
                          const float* __restrict__ kvb,
                          unsigned short* __restrict__ kt,
                          unsigned short* __restrict__ v) {
  int wave = (int)((blockIdx.x * 256 + threadIdx.x) >> 5);
  int lane = threadIdx.x & 31;
  int half = lane >> 4, sub = lane & 15;
  int nt = wave & 255, ot = (wave >> 8) & 31, b = wave >> 13;
  int o0 = ot * 16, n0 = nt * 16;
  floatx8 acc;
  #pragma unroll
  for (int i = 0; i < 8; ++i) acc[i] = 0.f;
  const unsigned short* arow = wkv + (size_t)(o0 + sub) * CC;
  const unsigned short* bcol = xt + ((size_t)b * NN + n0 + sub) * CC;
  #pragma unroll
  for (int kk = 0; kk < CC; kk += 32)
    acc = wmma_bf16(load_frag_a(arow + kk, half), load_frag_b(bcol + kk, half), acc);
  int n = n0 + sub;
  if (o0 < CC) {
    union { unsigned short s[8]; uintx4 u; } ou;
    #pragma unroll
    for (int r = 0; r < 8; ++r) ou.s[r] = f32_to_bf16(acc[r] + kvb[o0 + 8 * half + r]);
    *reinterpret_cast<uintx4*>(kt + ((size_t)b * NN + n) * CC + o0 + 8 * half) = ou.u;
  } else {
    #pragma unroll
    for (int r = 0; r < 8; ++r) {
      int o = o0 + 8 * half + r;
      int c = o - CC;
      v[((size_t)b * CC + c) * NN + n] = f32_to_bf16(acc[r] + kvb[o]);
    }
  }
}

// ---------------------------------------------------------------------------
// Flash attention: one wave = 16 query rows x 256 channels; the 8 waves of a
// block share double-buffered K/V tiles staged to LDS with async copies.
// ot_out (B,N,C) f32 = softmax(Q^T K / 16) V^T  (transposed layout)
__global__ void __launch_bounds__(256) k_attn(const unsigned short* __restrict__ qt,
                                              const unsigned short* __restrict__ kt,
                                              const unsigned short* __restrict__ v,
                                              float* __restrict__ ot_out) {
  __shared__ unsigned short lds_k[2][32 * CC];   // 16 KB per buffer (m-rows x C), contiguous slab of k_t
  __shared__ unsigned short lds_v[2][CC * 32];   // 16 KB per buffer (c-rows x m)
  __shared__ unsigned short pbuf[8][16 * 32];    // per-wave P transpose staging
  int tid  = threadIdx.x;
  int warp = tid >> 5;
  int lane = tid & 31;
  int half = lane >> 4, sub = lane & 15;
  int t = blockIdx.x * 8 + warp;
  int b = t >> 8;                 // constant across the block (32 blocks per batch)
  int n0 = (t & 255) * 16;

  const unsigned short* kbatch = kt + (size_t)b * NN * CC;
  const unsigned short* vbatch = v + (size_t)b * CC * NN;

  // block-cooperative stage of one 32-wide m-slice of K and V into LDS buffer `buf`
  auto stage = [&](int buf, int m0) {
    const unsigned short* ksrc = kbatch + (size_t)m0 * CC;   // 32*CC contiguous bf16
    const unsigned short* vsrc = vbatch + m0;
    #pragma unroll
    for (int j = 0; j < 4; ++j) {
      int ck = tid + 256 * j;                                // 0..1023 16B chunks
      cp16_to_lds(&lds_k[buf][ck * 8], ksrc + ck * 8);
      cp16_to_lds(&lds_v[buf][ck * 8], vsrc + (size_t)(ck >> 2) * NN + (ck & 3) * 8);
    }
  };

  bf16x16 qf[8];
  const unsigned short* qrow = qt + ((size_t)b * NN + n0 + sub) * CC;
  #pragma unroll
  for (int i = 0; i < 8; ++i) qf[i] = load_frag_a(qrow + 32 * i, half);

  floatx8 oacc[16];
  #pragma unroll
  for (int c = 0; c < 16; ++c)
    #pragma unroll
    for (int r = 0; r < 8; ++r) oacc[c][r] = 0.f;
  float rmax[8], rsum[8];
  #pragma unroll
  for (int r = 0; r < 8; ++r) { rmax[r] = -3.4e38f; rsum[r] = 0.f; }

  unsigned short* pw = &pbuf[warp][0];

  stage(0, 0);
  for (int m0 = 0; m0 < NN; m0 += 32) {
    int buf = (m0 >> 5) & 1;
    if (m0 + 32 < NN) {
      stage(buf ^ 1, m0 + 32);
      WAIT_ASYNC(8);              // in-order completion: oldest 8 (current buffer) done
    } else {
      WAIT_ASYNC(0);
    }
    __syncthreads();              // publish LDS tiles across waves

    const unsigned short* kl = &lds_k[buf][0];
    const unsigned short* vl = &lds_v[buf][0];

    floatx8 s0, s1;
    #pragma unroll
    for (int r = 0; r < 8; ++r) { s0[r] = 0.f; s1[r] = 0.f; }
    #pragma unroll
    for (int i = 0; i < 8; ++i) {
      s0 = wmma_bf16(qf[i], load_frag_b(kl + (size_t)sub * CC + 32 * i, half), s0);
      s1 = wmma_bf16(qf[i], load_frag_b(kl + (size_t)(16 + sub) * CC + 32 * i, half), s1);
    }
    s0 = s0 * 0.0625f;            // 1/sqrt(256)
    s1 = s1 * 0.0625f;

    float nm[8], f[8], rs[8];
    #pragma unroll
    for (int r = 0; r < 8; ++r) nm[r] = fmaxf(rmax[r], fmaxf(s0[r], s1[r]));
    #pragma unroll
    for (int r = 0; r < 8; ++r) {
      nm[r] = fmaxf(nm[r], __shfl_xor(nm[r], 1, 32));
      nm[r] = fmaxf(nm[r], __shfl_xor(nm[r], 2, 32));
      nm[r] = fmaxf(nm[r], __shfl_xor(nm[r], 4, 32));
      nm[r] = fmaxf(nm[r], __shfl_xor(nm[r], 8, 32));
    }
    #pragma unroll
    for (int r = 0; r < 8; ++r) {
      f[r] = __expf(rmax[r] - nm[r]);
      rmax[r] = nm[r];
      float p0 = __expf(s0[r] - nm[r]);
      float p1 = __expf(s1[r] - nm[r]);
      rs[r] = p0 + p1;
      pw[(8 * half + r) * 32 + sub]      = f32_to_bf16(p0);
      pw[(8 * half + r) * 32 + 16 + sub] = f32_to_bf16(p1);
    }
    #pragma unroll
    for (int r = 0; r < 8; ++r) {
      rs[r] += __shfl_xor(rs[r], 1, 32);
      rs[r] += __shfl_xor(rs[r], 2, 32);
      rs[r] += __shfl_xor(rs[r], 4, 32);
      rs[r] += __shfl_xor(rs[r], 8, 32);
      rsum[r] = rsum[r] * f[r] + rs[r];
    }
    #pragma unroll
    for (int c = 0; c < 16; ++c)
      #pragma unroll
      for (int r = 0; r < 8; ++r) oacc[c][r] *= f[r];

    WAIT_DS0();                   // P-tile cross-lane LDS dependence within the wave
    bf16x16 pf = load_frag_a(pw + sub * 32, half);   // P as A-fragment (n x m)
    #pragma unroll
    for (int c = 0; c < 16; ++c)
      oacc[c] = wmma_bf16(pf, load_frag_b(vl + (size_t)(c * 16 + sub) * 32, half), oacc[c]);

    __syncthreads();              // all waves done with buf before it is overwritten
  }

  #pragma unroll
  for (int r = 0; r < 8; ++r) rsum[r] = 1.f / rsum[r];
  int nrow = n0 + 8 * half;
  #pragma unroll
  for (int c = 0; c < 16; ++c)
    #pragma unroll
    for (int r = 0; r < 8; ++r)
      ot_out[((size_t)b * NN + nrow + r) * CC + c * 16 + sub] = oacc[c][r] * rsum[r];
}

// ---------------------------------------------------------------------------
// fused_t (B,N,C) bf16 = local(B,C,N) + gate[b] * attn_t(B,N,C)
__global__ void k_fuse(const float* __restrict__ local, const float* __restrict__ at,
                       const float* __restrict__ gate, unsigned short* __restrict__ fused) {
  size_t idx = (size_t)blockIdx.x * 256 + threadIdx.x;   // over B*N*C (c fastest)
  int c = (int)(idx & (CC - 1));
  size_t bn = idx >> 8;
  int b = (int)(bn >> 12);
  int n = (int)(bn & (NN - 1));
  float lv = local[((size_t)b * CC + c) * NN + n];
  fused[idx] = f32_to_bf16(lv + gate[b] * at[idx]);
}

// Output projection: d_out (B,C,N) f32 = Wo (C,C) @ fused (C,N) + b
__global__ void k_proj_out(const unsigned short* __restrict__ fused,
                           const unsigned short* __restrict__ wo,
                           const float* __restrict__ ob,
                           float* __restrict__ out) {
  int wave = (int)((blockIdx.x * 256 + threadIdx.x) >> 5);
  int lane = threadIdx.x & 31;
  int half = lane >> 4, sub = lane & 15;
  int nt = wave & 255, ot = (wave >> 8) & 15, b = wave >> 12;
  int o0 = ot * 16, n0 = nt * 16;
  floatx8 acc;
  #pragma unroll
  for (int i = 0; i < 8; ++i) acc[i] = 0.f;
  const unsigned short* arow = wo + (size_t)(o0 + sub) * CC;
  const unsigned short* bcol = fused + ((size_t)b * NN + n0 + sub) * CC;
  #pragma unroll
  for (int kk = 0; kk < CC; kk += 32)
    acc = wmma_bf16(load_frag_a(arow + kk, half), load_frag_b(bcol + kk, half), acc);
  int n = n0 + sub;
  #pragma unroll
  for (int r = 0; r < 8; ++r) {
    int o = o0 + 8 * half + r;
    out[((size_t)b * CC + o) * NN + n] = acc[r] + ob[o];
  }
}

// ---------------------------------------------------------------------------
extern "C" void kernel_launch(void* const* d_in, const int* in_sizes, int n_in,
                              void* d_out, int out_size, void* d_ws, size_t ws_size,
                              hipStream_t stream) {
  (void)in_sizes; (void)n_in; (void)out_size; (void)ws_size;

  const float* x_self = (const float*)d_in[0];
  const float* x_other = (const float*)d_in[1];
  const float* q_w  = (const float*)d_in[2];
  const float* q_b  = (const float*)d_in[3];
  const float* kv_w = (const float*)d_in[4];
  const float* kv_b = (const float*)d_in[5];
  const float* g1_w = (const float*)d_in[6];
  const float* g1_b = (const float*)d_in[7];
  const float* g2_w = (const float*)d_in[8];
  const float* g2_b = (const float*)d_in[9];
  const float* dw_w = (const float*)d_in[10];
  const float* dw_b = (const float*)d_in[11];
  const float* bn_g = (const float*)d_in[12];
  const float* bn_b = (const float*)d_in[13];
  const float* bn_m = (const float*)d_in[14];
  const float* bn_v = (const float*)d_in[15];
  const float* out_w = (const float*)d_in[16];
  const float* out_b = (const float*)d_in[17];
  float* out = (float*)d_out;

  char* w = (char*)d_ws;
  auto alloc = [&](size_t bytes) { char* p = w; w += (bytes + 255) & ~(size_t)255; return p; };
  const size_t BNC = (size_t)BB * NN * CC;

  unsigned short* xs_t  = (unsigned short*)alloc(BNC * 2);
  unsigned short* xo_t  = (unsigned short*)alloc(BNC * 2);
  unsigned short* q_t   = (unsigned short*)alloc(BNC * 2);
  unsigned short* k_t   = (unsigned short*)alloc(BNC * 2);
  unsigned short* v_b   = (unsigned short*)alloc(BNC * 2);
  float*          at_t  = (float*)alloc(BNC * 4);
  float*          local = (float*)alloc(BNC * 4);
  unsigned short* fused = (unsigned short*)alloc(BNC * 2);
  unsigned short* qw_bf = (unsigned short*)alloc((size_t)CC * CC * 2);
  unsigned short* kvw_bf= (unsigned short*)alloc((size_t)2 * CC * CC * 2);
  unsigned short* ow_bf = (unsigned short*)alloc((size_t)CC * CC * 2);
  float*          gate  = (float*)alloc(BB * 4);

  // 1) transpose-convert activations; convert weights to bf16
  k_transpose_bf16<<<dim3(CC / 32, NN / 32, BB), dim3(32, 8), 0, stream>>>(x_self, xs_t);
  k_transpose_bf16<<<dim3(CC / 32, NN / 32, BB), dim3(32, 8), 0, stream>>>(x_other, xo_t);
  k_convert_bf16<<<(CC * CC + 255) / 256, 256, 0, stream>>>(q_w, qw_bf, CC * CC);
  k_convert_bf16<<<(2 * CC * CC + 255) / 256, 256, 0, stream>>>(kv_w, kvw_bf, 2 * CC * CC);
  k_convert_bf16<<<(CC * CC + 255) / 256, 256, 0, stream>>>(out_w, ow_bf, CC * CC);

  // 2) gate scalar + local feature path
  k_gate<<<BB, CC, 0, stream>>>(x_self, g1_w, g1_b, g2_w, g2_b, gate);
  k_dwconv<<<(int)(BNC / 256), 256, 0, stream>>>(x_self, dw_w, dw_b, bn_g, bn_b, bn_m, bn_v, local);

  // 3) WMMA projections
  k_proj_q<<<(BB * (CC / 16) * (NN / 16)) / 8, 256, 0, stream>>>(xs_t, qw_bf, q_b, q_t);
  k_proj_kv<<<(BB * (2 * CC / 16) * (NN / 16)) / 8, 256, 0, stream>>>(xo_t, kvw_bf, kv_b, k_t, v_b);

  // 4) flash attention (async LDS staging + WMMA QK^T + online softmax + WMMA PV)
  k_attn<<<(BB * (NN / 16)) / 8, 256, 0, stream>>>(q_t, k_t, v_b, at_t);

  // 5) fuse + output projection
  k_fuse<<<(int)(BNC / 256), 256, 0, stream>>>(local, at_t, gate, fused);
  k_proj_out<<<(BB * (CC / 16) * (NN / 16)) / 8, 256, 0, stream>>>(fused, ow_bf, out_b, out);
}
